// Qwen3Attention_55929064128770
// MI455X (gfx1250) — matrix operand order repaired and checked by reading, not compiled, not run on hardware
//
#include <hip/hip_runtime.h>
#include <hip/hip_bf16.h>

// ---------------------------------------------------------------------------
// Qwen3 attention prefill for gfx1250 (MI455X), WMMA f16 path, f32 accumulate.
// B=2, T=2048, C=1024, 16 q-heads, 8 kv-heads, D=128.
// ---------------------------------------------------------------------------

typedef _Float16 h8  __attribute__((ext_vector_type(8)));
typedef _Float16 h16 __attribute__((ext_vector_type(16)));
typedef float    f8  __attribute__((ext_vector_type(8)));

#define WMMA_F16(a, b, c) \
  __builtin_amdgcn_wmma_f32_16x16x32_f16(false, (a), false, (b), (short)0, (c), false, false)

__device__ __forceinline__ h16 cat16(h8 lo, h8 hi) {
  return __builtin_shufflevector(lo, hi, 0,1,2,3,4,5,6,7,8,9,10,11,12,13,14,15);
}

// -------------------------------- constants --------------------------------
#define BB 2
#define TT 2048
#define CC 1024
#define HKV 8
#define HQ 16
#define DD 128
#define MM (BB*TT)            // 4096 rows of activations
#define SCALE 0.08838834764831845f   // 1/sqrt(128)

// ------------------------------ f32 -> f16 ---------------------------------
__global__ void __launch_bounds__(256) cvt_f16_kernel(
    const float* __restrict__ a, _Float16* __restrict__ o, int n) {
  int i = blockIdx.x * blockDim.x + threadIdx.x;
  if (i < n) o[i] = (_Float16)a[i];
}

// ------------------------- generic WMMA GEMM -------------------------------
// Y[M,N] = A[M,K] (f16, row-major) @ W[N,K]^T (f16, row-major rows = N).
// One wave computes a 32x32 tile with 2x2 wmma accumulators.
template <typename OutT>
__global__ void __launch_bounds__(256, 1) gemm_wmma_kernel(
    const _Float16* __restrict__ A,
    const _Float16* __restrict__ W,
    OutT* __restrict__ Y,
    int M, int N, int K) {
  int wave = (int)((blockIdx.x * blockDim.x + threadIdx.x) >> 5);
  int ntiles = N >> 5;
  int mt = wave / ntiles;
  int nt = wave - mt * ntiles;
  if (mt >= (M >> 5)) return;
  int lane = threadIdx.x & 31;
  int l16  = lane & 15;
  int hi   = lane >> 4;
  int m0 = mt << 5, n0 = nt << 5;

  f8 c00{}, c01{}, c10{}, c11{};

  const _Float16* pa0 = A + (size_t)(m0 + l16) * K + hi * 16;
  const _Float16* pa1 = pa0 + (size_t)16 * K;
  const _Float16* pb0 = W + (size_t)(n0 + l16) * K + hi * 16;
  const _Float16* pb1 = pb0 + (size_t)16 * K;

  for (int k = 0; k < K; k += 32) {
    h16 a0 = cat16(*(const h8*)(pa0 + k), *(const h8*)(pa0 + k + 8));
    h16 a1 = cat16(*(const h8*)(pa1 + k), *(const h8*)(pa1 + k + 8));
    h16 b0 = *(const h16*)(pb0 + k);
    h16 b1 = *(const h16*)(pb1 + k);
    c00 = WMMA_F16(a0, b0, c00);
    c01 = WMMA_F16(a0, b1, c01);
    c10 = WMMA_F16(a1, b0, c10);
    c11 = WMMA_F16(a1, b1, c11);
  }

  int col = n0 + l16;
#pragma unroll
  for (int r = 0; r < 8; ++r) {
    int row = m0 + r + hi * 8;
    Y[(size_t)row * N + col]             = (OutT)c00[r];
    Y[(size_t)row * N + col + 16]        = (OutT)c01[r];
    Y[(size_t)(row + 16) * N + col]      = (OutT)c10[r];
    Y[(size_t)(row + 16) * N + col + 16] = (OutT)c11[r];
  }
}

// ---------------------- RMSNorm + RoPE for Q -------------------------------
// One wave per (b, t, h) row of 128. Lane handles d = lane, +32, +64, +96.
__global__ void __launch_bounds__(256) rope_q_kernel(
    const _Float16* __restrict__ qraw,  // [B,T,2048]
    const float* __restrict__ cosT,     // [T,64]
    const float* __restrict__ sinT,
    const float* __restrict__ qw,       // [128]
    _Float16* __restrict__ q16) {       // [B,HQ,T,D]
  int wid = (int)((blockIdx.x * blockDim.x + threadIdx.x) >> 5);
  if (wid >= BB * TT * HQ) return;
  int lane = threadIdx.x & 31;
  int h = wid & (HQ - 1);
  int t = (wid >> 4) & (TT - 1);
  int b = wid >> 15;

  const _Float16* src = qraw + ((size_t)(b * TT + t)) * (HQ * DD) + h * DD;
  float x0 = (float)src[lane];
  float x1 = (float)src[lane + 32];
  float x2 = (float)src[lane + 64];
  float x3 = (float)src[lane + 96];
  float ss = x0*x0 + x1*x1 + x2*x2 + x3*x3;
#pragma unroll
  for (int off = 16; off; off >>= 1) ss += __shfl_xor(ss, off, 32);
  float inv = rsqrtf(ss * (1.0f / 128.0f) + 1e-6f);
  float n0 = x0 * inv * qw[lane];
  float n1 = x1 * inv * qw[lane + 32];
  float n2 = x2 * inv * qw[lane + 64];
  float n3 = x3 * inv * qw[lane + 96];
  float c0 = cosT[t * 64 + lane],      c1 = cosT[t * 64 + lane + 32];
  float s0 = sinT[t * 64 + lane],      s1 = sinT[t * 64 + lane + 32];

  _Float16* dst = q16 + (((size_t)(b * HQ + h)) * TT + t) * DD;
  dst[lane]      = (_Float16)(n0 * c0 - n2 * s0);
  dst[lane + 32] = (_Float16)(n1 * c1 - n3 * s1);
  dst[lane + 64] = (_Float16)(n0 * s0 + n2 * c0);
  dst[lane + 96] = (_Float16)(n1 * s1 + n3 * c1);
}

// ---------------------- RMSNorm + RoPE for K (+ new_k out) -----------------
__global__ void __launch_bounds__(256) rope_k_kernel(
    const _Float16* __restrict__ kraw,  // [B,T,1024]
    const float* __restrict__ cosT,
    const float* __restrict__ sinT,
    const float* __restrict__ kw,
    _Float16* __restrict__ k16,         // [B,HKV,T,D]
    float* __restrict__ outK) {         // [B,HKV,T,D]
  int wid = (int)((blockIdx.x * blockDim.x + threadIdx.x) >> 5);
  if (wid >= BB * TT * HKV) return;
  int lane = threadIdx.x & 31;
  int h = wid & (HKV - 1);
  int t = (wid >> 3) & (TT - 1);
  int b = wid >> 14;

  const _Float16* src = kraw + ((size_t)(b * TT + t)) * (HKV * DD) + h * DD;
  float x0 = (float)src[lane];
  float x1 = (float)src[lane + 32];
  float x2 = (float)src[lane + 64];
  float x3 = (float)src[lane + 96];
  float ss = x0*x0 + x1*x1 + x2*x2 + x3*x3;
#pragma unroll
  for (int off = 16; off; off >>= 1) ss += __shfl_xor(ss, off, 32);
  float inv = rsqrtf(ss * (1.0f / 128.0f) + 1e-6f);
  float n0 = x0 * inv * kw[lane];
  float n1 = x1 * inv * kw[lane + 32];
  float n2 = x2 * inv * kw[lane + 64];
  float n3 = x3 * inv * kw[lane + 96];
  float c0 = cosT[t * 64 + lane],      c1 = cosT[t * 64 + lane + 32];
  float s0 = sinT[t * 64 + lane],      s1 = sinT[t * 64 + lane + 32];

  float r0 = n0 * c0 - n2 * s0;
  float r1 = n1 * c1 - n3 * s1;
  float r2 = n0 * s0 + n2 * c0;
  float r3 = n1 * s1 + n3 * c1;

  size_t base = (((size_t)(b * HKV + h)) * TT + t) * DD;
  k16[base + lane]      = (_Float16)r0;
  k16[base + lane + 32] = (_Float16)r1;
  k16[base + lane + 64] = (_Float16)r2;
  k16[base + lane + 96] = (_Float16)r3;
  outK[base + lane]      = r0;
  outK[base + lane + 32] = r1;
  outK[base + lane + 64] = r2;
  outK[base + lane + 96] = r3;
}

// ------------------ V rearrange: [B,T,H,D] -> [B,H,D,T] + new_v ------------
__global__ void __launch_bounds__(256) v_pack_kernel(
    const _Float16* __restrict__ vraw,  // [B,T,1024]
    _Float16* __restrict__ vT16,        // [B,HKV,D,T]
    float* __restrict__ outV) {         // [B,HKV,T,D]
  int wid = (int)((blockIdx.x * blockDim.x + threadIdx.x) >> 5);
  if (wid >= BB * TT * HKV) return;
  int lane = threadIdx.x & 31;
  int h = wid & (HKV - 1);
  int t = (wid >> 3) & (TT - 1);
  int b = wid >> 14;

  const _Float16* src = vraw + ((size_t)(b * TT + t)) * (HKV * DD) + h * DD;
  size_t obase = (((size_t)(b * HKV + h)) * TT + t) * DD;
  size_t tbase = ((size_t)(b * HKV + h)) * DD;  // then (+d)*T + t
#pragma unroll
  for (int j = 0; j < 4; ++j) {
    int d = lane + j * 32;
    _Float16 v = src[d];
    vT16[(tbase + d) * TT + t] = v;
    outV[obase + d] = (float)v;
  }
}

// ------------------------- flash attention (wave32) ------------------------
// One wave per (b, q-head, 16-row query tile). Online softmax, f32 state.
// __launch_bounds__(32, 1): single wave per WG, allow the full VGPR budget so
// the Q fragments + 8 O accumulators stay resident (no scratch spills).
__global__ void __launch_bounds__(32, 1) attn_kernel(
    const _Float16* __restrict__ q16,   // [B,HQ,T,D]
    const _Float16* __restrict__ k16,   // [B,HKV,T,D]
    const _Float16* __restrict__ vT16,  // [B,HKV,D,T]
    _Float16* __restrict__ o16) {       // [B,T,2048]
  __shared__ _Float16 ldsP[16 * 32];

  int id = blockIdx.x;
  int qt = id & ((TT / 16) - 1);       // 0..127
  int h  = (id >> 7) & (HQ - 1);
  int b  = id >> 11;
  int hkv = h >> 1;                    // repeat_interleave(2): 2 q-heads per kv-head
  int qb = qt * 16;

  int lane = threadIdx.x & 31;
  int l16  = lane & 15;
  int hi   = lane >> 4;

  // Preload the 16x128 Q tile as four 16x32 A fragments.
  h16 aQ[4];
  const _Float16* qp =
      q16 + (((size_t)(b * HQ + h)) * TT + qb + l16) * DD + hi * 16;
#pragma unroll
  for (int dk = 0; dk < 4; ++dk)
    aQ[dk] = cat16(*(const h8*)(qp + dk * 32), *(const h8*)(qp + dk * 32 + 8));

  float m[8], l[8];
  f8 o[8];
#pragma unroll
  for (int r = 0; r < 8; ++r) { m[r] = -1e30f; l[r] = 0.0f; }
#pragma unroll
  for (int n = 0; n < 8; ++n) o[n] = f8{};

  int nkb = (qb + 16 + 31) >> 5;       // causal key-block count (32 keys/block)
  for (int kb = 0; kb < nkb; ++kb) {
    // S = Q K^T for 32 keys -> two 16x16 tiles.
    f8 s0{}, s1{};
    const _Float16* kp =
        k16 + (((size_t)(b * HKV + hkv)) * TT + kb * 32 + l16) * DD + hi * 16;
#pragma unroll
    for (int dk = 0; dk < 4; ++dk) {
      h16 b0 = *(const h16*)(kp + dk * 32);
      h16 b1 = *(const h16*)(kp + (size_t)16 * DD + dk * 32);
      s0 = WMMA_F16(aQ[dk], b0, s0);
      s1 = WMMA_F16(aQ[dk], b1, s1);
    }

    // Online softmax with causal mask; rescale O accumulators in-place.
    int c0 = kb * 32 + l16;
    int c1 = c0 + 16;
#pragma unroll
    for (int r = 0; r < 8; ++r) {
      int row = qb + r + hi * 8;
      float x0 = (c0 <= row) ? s0[r] * SCALE : -1e30f;
      float x1 = (c1 <= row) ? s1[r] * SCALE : -1e30f;
      float mx = fmaxf(x0, x1);
#pragma unroll
      for (int off = 8; off; off >>= 1) mx = fmaxf(mx, __shfl_xor(mx, off, 16));
      float mn = fmaxf(m[r], mx);
      float p0 = __expf(x0 - mn);
      float p1 = __expf(x1 - mn);
      float a  = __expf(m[r] - mn);
      float ps = p0 + p1;
#pragma unroll
      for (int off = 8; off; off >>= 1) ps += __shfl_xor(ps, off, 16);
      l[r] = l[r] * a + ps;
      m[r] = mn;
#pragma unroll
      for (int n = 0; n < 8; ++n) o[n][r] *= a;
      int rl = r + hi * 8;
      ldsP[rl * 32 + l16]      = (_Float16)p0;
      ldsP[rl * 32 + l16 + 16] = (_Float16)p1;
    }

    __syncthreads();  // LDS P tile visible (C-layout -> A-layout bounce)

    const _Float16* pp = &ldsP[l16 * 32 + hi * 16];
    h16 aP = cat16(*(const h8*)pp, *(const h8*)(pp + 8));

    // O += P V : eight 16x16 d-tiles, V^T rows are contiguous in key dim.
    const _Float16* vp =
        vT16 + (((size_t)(b * HKV + hkv)) * DD + l16) * TT + kb * 32 + hi * 16;
#pragma unroll
    for (int n = 0; n < 8; ++n) {
      h16 bv = *(const h16*)(vp + (size_t)(n * 16) * TT);
      o[n] = WMMA_F16(aP, bv, o[n]);
    }
    __syncthreads();
  }

  // Normalize and store into [B,T,2048] f16 for the output projection.
#pragma unroll
  for (int r = 0; r < 8; ++r) {
    float inv = 1.0f / l[r];
    int row = qb + r + hi * 8;
    _Float16* op = o16 + ((size_t)(b * TT) + row) * (HQ * DD) + h * DD + l16;
#pragma unroll
    for (int n = 0; n < 8; ++n) op[n * 16] = (_Float16)(o[n][r] * inv);
  }
}

// ------------------------------- launcher ----------------------------------
extern "C" void kernel_launch(void* const* d_in, const int* in_sizes, int n_in,
                              void* d_out, int out_size, void* d_ws, size_t ws_size,
                              hipStream_t stream) {
  const float* x    = (const float*)d_in[0];
  const float* cosT = (const float*)d_in[1];
  const float* sinT = (const float*)d_in[2];
  const float* wq   = (const float*)d_in[3];
  const float* wk   = (const float*)d_in[4];
  const float* wv   = (const float*)d_in[5];
  const float* wo   = (const float*)d_in[6];
  const float* qnw  = (const float*)d_in[7];
  const float* knw  = (const float*)d_in[8];

  float* outY = (float*)d_out;                      // [B,T,C]
  float* outK = outY + (size_t)BB * TT * CC;        // [B,HKV,T,D]
  float* outV = outK + (size_t)BB * HKV * TT * DD;  // [B,HKV,T,D]

  char* ws = (char*)d_ws;
  _Float16* x16  = (_Float16*)ws; ws += (size_t)MM * CC * 2;            // 8 MB
  _Float16* wq16 = (_Float16*)ws; ws += (size_t)(2*CC) * CC * 2;        // 4 MB
  _Float16* wk16 = (_Float16*)ws; ws += (size_t)CC * CC * 2;            // 2 MB
  _Float16* wv16 = (_Float16*)ws; ws += (size_t)CC * CC * 2;            // 2 MB
  _Float16* wo16 = (_Float16*)ws; ws += (size_t)CC * (2*CC) * 2;        // 4 MB
  _Float16* qraw = (_Float16*)ws; ws += (size_t)MM * (2*CC) * 2;        // 16 MB
  _Float16* kraw = (_Float16*)ws; ws += (size_t)MM * CC * 2;            // 8 MB
  _Float16* vraw = (_Float16*)ws; ws += (size_t)MM * CC * 2;            // 8 MB
  _Float16* q16  = (_Float16*)ws; ws += (size_t)BB*HQ*TT*DD * 2;        // 16 MB
  _Float16* k16  = (_Float16*)ws; ws += (size_t)BB*HKV*TT*DD * 2;       // 8 MB
  _Float16* vT16 = (_Float16*)ws; ws += (size_t)BB*HKV*DD*TT * 2;       // 8 MB
  _Float16* o16  = (_Float16*)ws; ws += (size_t)MM * (2*CC) * 2;        // 16 MB

  // 1) f32 -> f16 conversions.
  auto cvt = [&](const float* a, _Float16* o, int n) {
    cvt_f16_kernel<<<(n + 255) / 256, 256, 0, stream>>>(a, o, n);
  };
  cvt(x,  x16,  MM * CC);
  cvt(wq, wq16, 2 * CC * CC);
  cvt(wk, wk16, CC * CC);
  cvt(wv, wv16, CC * CC);
  cvt(wo, wo16, CC * 2 * CC);

  // 2) QKV projections (WMMA).  waves = (M/32)*(N/32), 8 waves per block.
  {
    int waves_q = (MM / 32) * ((2 * CC) / 32);   // 8192
    gemm_wmma_kernel<_Float16><<<waves_q / 8, 256, 0, stream>>>(
        x16, wq16, qraw, MM, 2 * CC, CC);
    int waves_kv = (MM / 32) * (CC / 32);        // 4096
    gemm_wmma_kernel<_Float16><<<waves_kv / 8, 256, 0, stream>>>(
        x16, wk16, kraw, MM, CC, CC);
    gemm_wmma_kernel<_Float16><<<waves_kv / 8, 256, 0, stream>>>(
        x16, wv16, vraw, MM, CC, CC);
  }

  // 3) RMSNorm + RoPE (Q, K) and V repack (+ new_k/new_v fp32 outputs).
  rope_q_kernel<<<(BB * TT * HQ) / 8, 256, 0, stream>>>(
      qraw, cosT, sinT, qnw, q16);
  rope_k_kernel<<<(BB * TT * HKV) / 8, 256, 0, stream>>>(
      kraw, cosT, sinT, knw, k16, outK);
  v_pack_kernel<<<(BB * TT * HKV) / 8, 256, 0, stream>>>(
      vraw, vT16, outV);

  // 4) Flash attention: one wave per (b, head, 16-row q tile).
  attn_kernel<<<BB * HQ * (TT / 16), 32, 0, stream>>>(q16, k16, vT16, o16);

  // 5) Output projection: d_out[:B*T*C] = o16 @ wo^T (fp32 store).
  {
    int waves_o = (MM / 32) * (CC / 32);         // 4096
    gemm_wmma_kernel<float><<<waves_o / 8, 256, 0, stream>>>(
        o16, wo16, outY, MM, CC, 2 * CC);
  }
}
